// AdvancedPassPredictionModel_35897336660024
// MI455X (gfx1250) — matrix operand order repaired and checked by reading, not compiled, not run on hardware
//
#include <hip/hip_runtime.h>
#include <hip/hip_bf16.h>

// ---------------------------------------------------------------------------
// AdvancedPassPredictionModel forward for MI455X (gfx1250, wave32, WMMA).
// bf16 inputs / f32 accumulation via v_wmma_f32_16x16x32_bf16.
// GEMM: double-buffered LDS pipeline fed by GLOBAL_LOAD_ASYNC_TO_LDS_B128
// (ASYNCcnt, in-order completion -> s_wait_asynccnt 3 overlaps copy w/ WMMA).
// Weights pre-transposed to [N][K] so both tiles are contiguous row copies.
// ---------------------------------------------------------------------------

#define GEMM_USE_ASYNC 1

typedef __bf16 bf16;
typedef __attribute__((ext_vector_type(16))) __bf16 v16bf;
typedef __attribute__((ext_vector_type(8)))  float  v8f;

union FragBF { v16bf v; uint4 q[2]; };

__device__ inline v8f zero8() {
  v8f z = {0.f, 0.f, 0.f, 0.f, 0.f, 0.f, 0.f, 0.f};
  return z;
}

__device__ inline v8f wmma_bf16(v16bf a, v16bf b, v8f c) {
  // (neg_a, A, neg_b, B, c_mod, C, reuse_a, reuse_b)
  return __builtin_amdgcn_wmma_f32_16x16x32_bf16(false, a, false, b, (short)0, c,
                                                 false, false);
}

// Per-lane async copy of 16 bytes global -> LDS (VFLAT async, ASYNCcnt).
__device__ inline void async_copy_b128(void* lds, const void* gptr) {
  uint32_t lo = (uint32_t)(uintptr_t)lds;  // generic LDS ptr low 32 = LDS offset
  unsigned long long ga = (unsigned long long)(uintptr_t)gptr;
  asm volatile("global_load_async_to_lds_b128 %0, %1, off"
               :: "v"(lo), "v"(ga) : "memory");
}
// Wait until at most N async ops outstanding (in-order completion per wave).
__device__ inline void wait_async0() {
  asm volatile("s_wait_asynccnt 0x0" ::: "memory");
}
__device__ inline void wait_async3() {
  asm volatile("s_wait_asynccnt 0x3" ::: "memory");
}

// A-fragment: 16x32 bf16, row-major source, row stride ld (elements).
// lanes 0-15 -> row=lane, elems[0..7]=K0..7, [8..15]=K16..23
// lanes 16-31 -> row=lane-16, K8..15 / K24..31
__device__ inline v16bf load_a_frag(const bf16* base, int ld) {
  int l = threadIdx.x & 31;
  int r = l & 15;
  int koff = (l & 16) ? 8 : 0;
  FragBF f;
  const bf16* p = base + (long)r * ld;
  f.q[0] = *(const uint4*)(p + koff);
  f.q[1] = *(const uint4*)(p + koff + 16);
  return f.v;
}

// B-fragment: 32x16 (KxN) from [N][K] storage, row stride ld (elements).
// lanes 0-15: col=lane, K0..7 / K16..23; lanes 16-31: K8..15 / K24..31.
__device__ inline v16bf load_b_frag(const bf16* baseT, int ld) {
  int l = threadIdx.x & 31;
  int r = l & 15;
  int koff = (l & 16) ? 8 : 0;
  FragBF f;
  const uint4* p = (const uint4*)(baseT + (long)r * ld + koff);
  f.q[0] = p[0];
  f.q[1] = p[2];
  return f.v;
}

// ---------------------------------------------------------------------------
// fp32 -> bf16 conversions (plain and transposing: [K][N] f32 -> [N][K] bf16)
// ---------------------------------------------------------------------------
__global__ void cvt_f32_to_bf16(const float* __restrict__ in,
                                bf16* __restrict__ out, long n) {
  long i = (long)blockIdx.x * blockDim.x + threadIdx.x;
  if (i < n) out[i] = (bf16)in[i];
}

__global__ void cvt_f32_to_bf16_T(const float* __restrict__ in,
                                  bf16* __restrict__ out, int K, int N) {
  long i = (long)blockIdx.x * blockDim.x + threadIdx.x;
  if (i < (long)K * N) {
    int k = (int)(i / N), n = (int)(i % N);
    out[(long)n * K + k] = (bf16)in[i];
  }
}

// ---------------------------------------------------------------------------
// Row LayerNorm: fp32 in -> bf16 out.  One block (256 thr) per row.
// ---------------------------------------------------------------------------
__device__ inline float block_reduce_sum(float v, float* buf) {
  int t = threadIdx.x;
  buf[t] = v;
  __syncthreads();
  for (int o = 128; o > 0; o >>= 1) {
    if (t < o) buf[t] += buf[t + o];
    __syncthreads();
  }
  float r = buf[0];
  __syncthreads();
  return r;
}

__global__ __launch_bounds__(256)
void layernorm_to_bf16(const float* __restrict__ X, const float* __restrict__ g,
                       const float* __restrict__ bta, bf16* __restrict__ out,
                       int C) {
  __shared__ float rbuf[256];
  long row = blockIdx.x;
  const float* x = X + row * C;
  float s = 0.f, s2 = 0.f;
  for (int c = threadIdx.x; c < C; c += 256) {
    float v = x[c];
    s += v;
    s2 += v * v;
  }
  s = block_reduce_sum(s, rbuf);
  s2 = block_reduce_sum(s2, rbuf);
  float mean = s / C;
  float var = s2 / C - mean * mean;
  float inv = rsqrtf(var + 1e-5f);
  for (int c = threadIdx.x; c < C; c += 256)
    out[row * C + c] = (bf16)((x[c] - mean) * inv * g[c] + bta[c]);
}

// x2 = LayerNorm(x1 + gate * tproj), fp32 out. C==1024.
__global__ __launch_bounds__(256)
void gated_add_ln(const float* __restrict__ x1, const float* __restrict__ gate,
                  const float* __restrict__ tproj, const float* __restrict__ g,
                  const float* __restrict__ bta, float* __restrict__ out, int C) {
  __shared__ float rbuf[256];
  __shared__ float tbuf[1024];
  long row = blockIdx.x;
  float s = 0.f, s2 = 0.f;
  for (int c = threadIdx.x; c < C; c += 256) {
    float v = x1[row * C + c] + gate[row * C + c] * tproj[row * C + c];
    tbuf[c] = v;
    s += v;
    s2 += v * v;
  }
  __syncthreads();
  s = block_reduce_sum(s, rbuf);
  s2 = block_reduce_sum(s2, rbuf);
  float mean = s / C;
  float var = s2 / C - mean * mean;
  float inv = rsqrtf(var + 1e-5f);
  for (int c = threadIdx.x; c < C; c += 256)
    out[row * C + c] = (tbuf[c] - mean) * inv * g[c] + bta[c];
}

// ---------------------------------------------------------------------------
// pooled = mean over S  (X: [B,S,Dd] -> out [B,Dd])
// ---------------------------------------------------------------------------
template <typename T>
__global__ void pooled_mean(const T* __restrict__ X, float* __restrict__ out,
                            int S, int Dd) {
  int idx = blockIdx.x * blockDim.x + threadIdx.x;
  int b = idx / Dd, d = idx % Dd;
  float s = 0.f;
  for (int t = 0; t < S; ++t) s += (float)X[((long)b * S + t) * Dd + d];
  out[idx] = s / S;
}

// ---------------------------------------------------------------------------
// Head-importance MLP: imp = softmax(relu(pooled@w1+b1)@w2+b2). One block/batch.
// ---------------------------------------------------------------------------
__global__ __launch_bounds__(128)
void importance_kernel(const float* __restrict__ pooled,
                       const float* __restrict__ w1, const float* __restrict__ b1,
                       const float* __restrict__ w2, const float* __restrict__ b2,
                       float* __restrict__ imp, int Din, int Dh, int Hh) {
  __shared__ float hid[512];
  __shared__ float logit[16];
  int b = blockIdx.x;
  const float* pv = pooled + b * Din;
  for (int j = threadIdx.x; j < Dh; j += blockDim.x) {
    float s = b1[j];
    for (int i = 0; i < Din; ++i) s += pv[i] * w1[i * Dh + j];
    hid[j] = fmaxf(s, 0.f);
  }
  __syncthreads();
  if ((int)threadIdx.x < Hh) {
    float s = b2[threadIdx.x];
    for (int j = 0; j < Dh; ++j) s += hid[j] * w2[j * Hh + threadIdx.x];
    logit[threadIdx.x] = s;
  }
  __syncthreads();
  if (threadIdx.x == 0) {
    float m = -1e30f;
    for (int h = 0; h < Hh; ++h) m = fmaxf(m, logit[h]);
    float sum = 0.f;
    for (int h = 0; h < Hh; ++h) {
      float e = __expf(logit[h] - m);
      logit[h] = e;
      sum += e;
    }
    for (int h = 0; h < Hh; ++h) imp[b * Hh + h] = logit[h] / sum;
  }
}

// ---------------------------------------------------------------------------
// WMMA GEMM: C[M,N] = act(A[M,K](bf16) @ W[K,N] + bias + resid)
// W passed TRANSPOSED: WT is [N][K] bf16, row stride ldw.
// BM=128 BN=64 BK=32, 256 threads (8 wave32), wave w owns rows w*16..w*16+15.
// Depth-1 software pipeline: async-copy tile k+1 into the ping-pong LDS
// buffer, s_wait_asynccnt 3 (tile k arrived, k+1 in flight), barrier, WMMA.
// ACT: 0 none, 1 relu, 2 sigmoid.  Fully templated epilogue (no branches).
// ---------------------------------------------------------------------------
#define GBM 128
#define GBN 64
#define GBK 32
#define GSTR 40  // padded LDS row stride in elements (80B, 16B aligned)

template <int ACT, bool BIAS, bool RES, bool OUTF, bool OUTB>
__global__ __launch_bounds__(256)
void gemm_bf16(const bf16* __restrict__ A, const bf16* __restrict__ WT, int ldw,
               const float* __restrict__ bias, const float* __restrict__ resid,
               float* __restrict__ outF, bf16* __restrict__ outB,
               int M, int N, int K) {
  __shared__ __align__(16) bf16 As[2][GBM][GSTR];
  __shared__ __align__(16) bf16 Ws[2][GBN][GSTR];
  int tid = threadIdx.x;
  int wave = tid >> 5;
  int bm = blockIdx.y * GBM;
  int bn = blockIdx.x * GBN;
  v8f acc[4] = {zero8(), zero8(), zero8(), zero8()};

  int ar0 = tid >> 2;      // A row for this thread's chunks (also W n-row)
  int ac = (tid & 3) * 8;  // col offset 0,8,16,24

  const bf16* arow0 = A + (long)(bm + ar0) * K + ac;
  const bf16* arow1 = A + (long)(bm + ar0 + 64) * K + ac;
  const bf16* wrow = WT + (long)(bn + ar0) * ldw + ac;

  auto issue = [&](int t, int buf) {
    int kk = t * GBK;
#if GEMM_USE_ASYNC
    async_copy_b128(&As[buf][ar0][ac], arow0 + kk);
    async_copy_b128(&As[buf][ar0 + 64][ac], arow1 + kk);
    async_copy_b128(&Ws[buf][ar0][ac], wrow + kk);
#else
    *(uint4*)&As[buf][ar0][ac] = *(const uint4*)(arow0 + kk);
    *(uint4*)&As[buf][ar0 + 64][ac] = *(const uint4*)(arow1 + kk);
    *(uint4*)&Ws[buf][ar0][ac] = *(const uint4*)(wrow + kk);
#endif
  };

  int nk = K / GBK;
  issue(0, 0);
  for (int t = 0; t < nk; ++t) {
    int cur = t & 1;
    if (t + 1 < nk) {
      issue(t + 1, cur ^ 1);
      if (t + 2 < nk) {  // keep L2 warm ahead of the async engine
        __builtin_prefetch(arow0 + (t + 2) * GBK, 0, 1);
        __builtin_prefetch(wrow + (t + 2) * GBK, 0, 1);
      }
#if GEMM_USE_ASYNC
      wait_async3();  // tile t complete; tile t+1 still in flight
#endif
    } else {
#if GEMM_USE_ASYNC
      wait_async0();
#endif
    }
    __syncthreads();
    v16bf af = load_a_frag(&As[cur][wave * 16][0], GSTR);
#pragma unroll
    for (int nt = 0; nt < 4; ++nt) {
      v16bf bfrag = load_b_frag(&Ws[cur][nt * 16][0], GSTR);
      acc[nt] = wmma_bf16(af, bfrag, acc[nt]);
    }
    __syncthreads();  // everyone done reading buf cur before it is refilled
  }

  // Epilogue.  C layout: lane l, vgpr v -> M = v + (l>=16 ? 8:0), N = l&15.
  int l = tid & 31;
  int n_in = l & 15;
  int mbase = (l & 16) ? 8 : 0;
#pragma unroll
  for (int nt = 0; nt < 4; ++nt) {
    int col = bn + nt * 16 + n_in;
    float bv = BIAS ? bias[col] : 0.f;
#pragma unroll
    for (int v = 0; v < 8; ++v) {
      int row = bm + wave * 16 + mbase + v;
      float x = acc[nt][v] + bv;
      if (RES) x += resid[(long)row * N + col];
      if (ACT == 1) x = fmaxf(x, 0.f);
      if (ACT == 2) x = 1.f / (1.f + __expf(-x));
      if (OUTF) outF[(long)row * N + col] = x;
      if (OUTB) outB[(long)row * N + col] = (bf16)x;
    }
  }
}

// ---------------------------------------------------------------------------
// Attention per (b, h, 32-query-row tile).  Block = 64 threads (2 waves).
// Dynamic LDS: scores f32 [32][S] | P bf16 [32][S] | VT bf16 [DK][40] | red[64]
// ---------------------------------------------------------------------------
template <int DK>
__global__ __launch_bounds__(64)
void attn_flash(const bf16* __restrict__ Q, const bf16* __restrict__ Km,
                const bf16* __restrict__ V, const float* __restrict__ imp,
                bf16* __restrict__ ctx, int S, int H, float scale) {
  extern __shared__ char smem[];
  const int D = H * DK;
  float* sc = (float*)smem;                                        // 32*S f32
  bf16* P = (bf16*)(smem + (size_t)32 * S * 4);                    // 32*S bf16
  bf16* VT = (bf16*)(smem + (size_t)32 * S * 6);                   // DK*40 bf16
  float* red = (float*)(smem + (size_t)32 * S * 6 + DK * 40 * 2);  // 64 f32

  int tid = threadIdx.x;
  int wave = tid >> 5;
  int qb = blockIdx.x * 32;
  int h = blockIdx.y;
  int b = blockIdx.z;
  int l = tid & 31;
  int n_in = l & 15;
  int mbase = (l & 16) ? 8 : 0;

  v16bf aq[DK / 32];
#pragma unroll
  for (int i = 0; i < DK / 32; ++i)
    aq[i] = load_a_frag(Q + (long)(b * S + qb + wave * 16) * D + h * DK + i * 32, D);

  // Pass A: scores = scale * Q K^T  (K's row-major rows ARE the B-frag layout)
  for (int nt = 0; nt < S / 16; ++nt) {
    v8f acc = zero8();
#pragma unroll
    for (int i = 0; i < DK / 32; ++i) {
      v16bf bk = load_b_frag(Km + (long)(b * S + nt * 16) * D + h * DK + i * 32, D);
      acc = wmma_bf16(aq[i], bk, acc);
    }
#pragma unroll
    for (int v = 0; v < 8; ++v)
      sc[(wave * 16 + mbase + v) * S + nt * 16 + n_in] = acc[v] * scale;
  }
  __syncthreads();

  // Pass B: row softmax -> bf16 P.  Thread t: row t&31, half t>>5.
  {
    int r = tid & 31;
    int half = tid >> 5;
    float* rowp = sc + (long)r * S + half * (S / 2);
    float m = -1e30f;
    for (int c = 0; c < S / 2; ++c) m = fmaxf(m, rowp[c]);
    red[half * 32 + r] = m;
    __syncthreads();
    m = fmaxf(red[r], red[32 + r]);
    float sum = 0.f;
    for (int c = 0; c < S / 2; ++c) sum += __expf(rowp[c] - m);
    __syncthreads();
    red[half * 32 + r] = sum;
    __syncthreads();
    float inv = 1.f / (red[r] + red[32 + r]);
    bf16* prow = P + (long)r * S + half * (S / 2);
    for (int c = 0; c < S / 2; ++c) prow[c] = (bf16)(__expf(rowp[c] - m) * inv);
  }
  __syncthreads();

  // Pass C: O = P @ V, V staged transposed in LDS 32 keys at a time.
  constexpr int NT = DK / 16;
  v8f oacc[NT];
#pragma unroll
  for (int nt = 0; nt < NT; ++nt) oacc[nt] = zero8();

  for (int kk = 0; kk < S; kk += 32) {
    for (int i = tid; i < 32 * DK; i += 64) {
      int kr = i / DK;
      int d = i % DK;
      VT[d * 40 + kr] = V[(long)(b * S + kk + kr) * D + h * DK + d];
    }
    __syncthreads();
    v16bf ap = load_a_frag(P + (long)(wave * 16) * S + kk, S);
#pragma unroll
    for (int nt = 0; nt < NT; ++nt) {
      v16bf bv = load_b_frag(VT + nt * 16 * 40, 40);
      oacc[nt] = wmma_bf16(ap, bv, oacc[nt]);
    }
    __syncthreads();
  }

  float iw = imp[b * H + h];
#pragma unroll
  for (int nt = 0; nt < NT; ++nt)
#pragma unroll
    for (int v = 0; v < 8; ++v) {
      long row = (long)(b * S + qb + wave * 16 + mbase + v);
      ctx[row * D + h * DK + nt * 16 + n_in] = (bf16)(oacc[nt][v] * iw);
    }
}

// ---------------------------------------------------------------------------
// Host orchestration
// ---------------------------------------------------------------------------
extern "C" void kernel_launch(void* const* d_in, const int* in_sizes, int n_in,
                              void* d_out, int out_size, void* d_ws, size_t ws_size,
                              hipStream_t stream) {
  (void)in_sizes; (void)n_in; (void)out_size; (void)ws_size;
  const int Bb = 8, S = 1024, D = 1024, H = 16;
  const int TD = 256, THh = 8, FF = 4096;
  const int R = Bb * S;

  const float* x     = (const float*)d_in[0];
  const float* tc    = (const float*)d_in[1];
  const float* sa_wq = (const float*)d_in[2];
  const float* sa_wk = (const float*)d_in[3];
  const float* sa_wv = (const float*)d_in[4];
  const float* sa_iw1 = (const float*)d_in[5];
  const float* sa_ib1 = (const float*)d_in[6];
  const float* sa_iw2 = (const float*)d_in[7];
  const float* sa_ib2 = (const float*)d_in[8];
  const float* sa_ow = (const float*)d_in[9];
  const float* sa_ob = (const float*)d_in[10];
  const float* ta_wq = (const float*)d_in[11];
  const float* ta_wk = (const float*)d_in[12];
  const float* ta_wv = (const float*)d_in[13];
  const float* ta_iw1 = (const float*)d_in[14];
  const float* ta_ib1 = (const float*)d_in[15];
  const float* ta_iw2 = (const float*)d_in[16];
  const float* ta_ib2 = (const float*)d_in[17];
  const float* ta_ow = (const float*)d_in[18];
  const float* ta_ob = (const float*)d_in[19];
  const float* ff_w1 = (const float*)d_in[20];
  const float* ff_b1 = (const float*)d_in[21];
  const float* ff_w2 = (const float*)d_in[22];
  const float* ff_b2 = (const float*)d_in[23];
  const float* n1_g = (const float*)d_in[24];
  const float* n1_b = (const float*)d_in[25];
  const float* n2_g = (const float*)d_in[26];
  const float* n2_b = (const float*)d_in[27];
  const float* n3_g = (const float*)d_in[28];
  const float* n3_b = (const float*)d_in[29];
  const float* tp_w = (const float*)d_in[30];
  const float* tp_b = (const float*)d_in[31];
  const float* tg_w = (const float*)d_in[32];
  const float* tg_b = (const float*)d_in[33];

  size_t off = 0;
  auto alloc = [&](size_t bytes) -> void* {
    off = (off + 255) & ~(size_t)255;
    void* p = (char*)d_ws + off;
    off += bytes;
    return p;
  };
  // transposed bf16 weights [N][K]
  bf16* wqT = (bf16*)alloc((size_t)D * D * 2);
  bf16* wkT = (bf16*)alloc((size_t)D * D * 2);
  bf16* wvT = (bf16*)alloc((size_t)D * D * 2);
  bf16* owT = (bf16*)alloc((size_t)D * D * 2);
  bf16* twqT = (bf16*)alloc((size_t)TD * TD * 2);
  bf16* twkT = (bf16*)alloc((size_t)TD * TD * 2);
  bf16* twvT = (bf16*)alloc((size_t)TD * TD * 2);
  bf16* towT = (bf16*)alloc((size_t)TD * TD * 2);
  bf16* ffw1T = (bf16*)alloc((size_t)D * FF * 2);
  bf16* ffw2T = (bf16*)alloc((size_t)FF * D * 2);
  bf16* tpwT = (bf16*)alloc((size_t)TD * D * 2);
  bf16* tgwT = (bf16*)alloc((size_t)(D + TD) * D * 2);
  bf16* tc_bf = (bf16*)alloc((size_t)R * TD * 2);
  bf16* xn_bf = (bf16*)alloc((size_t)R * D * 2);  // xn1, later reused for xn3
  char* qkv_region = (char*)alloc((size_t)R * FF * 2);  // QKV, later FFN hidden
  bf16* Qbf = (bf16*)qkv_region;
  bf16* Kbf = (bf16*)(qkv_region + (size_t)R * D * 2);
  bf16* Vbf = (bf16*)(qkv_region + (size_t)R * D * 4);
  bf16* ffh_bf = (bf16*)qkv_region;
  bf16* ctx_bf = (bf16*)alloc((size_t)R * D * 2);
  bf16* Qt = (bf16*)alloc((size_t)R * TD * 2);
  bf16* Kt = (bf16*)alloc((size_t)R * TD * 2);
  bf16* Vt = (bf16*)alloc((size_t)R * TD * 2);
  bf16* ctxT_bf = (bf16*)alloc((size_t)R * TD * 2);
  float* pooled = (float*)alloc((size_t)Bb * D * 4);
  float* pooledT = (float*)alloc((size_t)Bb * TD * 4);
  float* impM = (float*)alloc((size_t)Bb * H * 4);
  float* impT = (float*)alloc((size_t)Bb * THh * 4);
  float* x1 = (float*)alloc((size_t)R * D * 4);
  bf16* x1_bf = (bf16*)alloc((size_t)R * D * 2);
  float* tf = (float*)alloc((size_t)R * TD * 4);
  bf16* tf_bf = (bf16*)alloc((size_t)R * TD * 2);
  float* tproj = (float*)alloc((size_t)R * D * 4);
  float* gate = (float*)alloc((size_t)R * D * 4);
  float* x2 = (float*)alloc((size_t)R * D * 4);

  auto cvtT = [&](const float* src, bf16* dst, int K, int N) {
    long n = (long)K * N;
    cvt_f32_to_bf16_T<<<(unsigned)((n + 255) / 256), 256, 0, stream>>>(src, dst, K, N);
  };

  // 1) weight conversions to transposed bf16; tc to bf16
  cvtT(sa_wq, wqT, D, D);
  cvtT(sa_wk, wkT, D, D);
  cvtT(sa_wv, wvT, D, D);
  cvtT(sa_ow, owT, D, D);
  cvtT(ta_wq, twqT, TD, TD);
  cvtT(ta_wk, twkT, TD, TD);
  cvtT(ta_wv, twvT, TD, TD);
  cvtT(ta_ow, towT, TD, TD);
  cvtT(ff_w1, ffw1T, D, FF);
  cvtT(ff_w2, ffw2T, FF, D);
  cvtT(tp_w, tpwT, TD, D);
  cvtT(tg_w, tgwT, D + TD, D);
  {
    long n = (long)R * TD;
    cvt_f32_to_bf16<<<(unsigned)((n + 255) / 256), 256, 0, stream>>>(tc, tc_bf, n);
  }

  // 2) LN1
  layernorm_to_bf16<<<R, 256, 0, stream>>>(x, n1_g, n1_b, xn_bf, D);

  // 3) pooled means + importance MLPs
  pooled_mean<bf16><<<(Bb * D) / 256, 256, 0, stream>>>(xn_bf, pooled, S, D);
  pooled_mean<float><<<(Bb * TD) / 256, 256, 0, stream>>>(tc, pooledT, S, TD);
  importance_kernel<<<Bb, 128, 0, stream>>>(pooled, sa_iw1, sa_ib1, sa_iw2,
                                            sa_ib2, impM, D, D / 2, H);
  importance_kernel<<<Bb, 128, 0, stream>>>(pooledT, ta_iw1, ta_ib1, ta_iw2,
                                            ta_ib2, impT, TD, TD / 2, THh);

#define GEMM(ACT, BIAS, RES, OUTF, OUTB, A_, WT_, ldw_, b_, r_, of_, ob_, N_, K_)   \
  do {                                                                              \
    dim3 g_((N_) / GBN, R / GBM);                                                   \
    gemm_bf16<ACT, BIAS, RES, OUTF, OUTB>                                           \
        <<<g_, 256, 0, stream>>>(A_, WT_, ldw_, b_, r_, of_, ob_, R, N_, K_);       \
  } while (0)

  // 4) QKV projections (main + tactical)
  GEMM(0, false, false, false, true, xn_bf, wqT, D, nullptr, nullptr, nullptr, Qbf, D, D);
  GEMM(0, false, false, false, true, xn_bf, wkT, D, nullptr, nullptr, nullptr, Kbf, D, D);
  GEMM(0, false, false, false, true, xn_bf, wvT, D, nullptr, nullptr, nullptr, Vbf, D, D);
  GEMM(0, false, false, false, true, tc_bf, twqT, TD, nullptr, nullptr, nullptr, Qt, TD, TD);
  GEMM(0, false, false, false, true, tc_bf, twkT, TD, nullptr, nullptr, nullptr, Kt, TD, TD);
  GEMM(0, false, false, false, true, tc_bf, twvT, TD, nullptr, nullptr, nullptr, Vt, TD, TD);

  // 5) attention
  size_t lds64 = (size_t)32 * S * 4 + (size_t)32 * S * 2 + 64 * 40 * 2 + 64 * 4;
  size_t lds32 = (size_t)32 * S * 4 + (size_t)32 * S * 2 + 32 * 40 * 2 + 64 * 4;
  (void)hipFuncSetAttribute(reinterpret_cast<const void*>(&attn_flash<64>),
                            hipFuncAttributeMaxDynamicSharedMemorySize, (int)lds64);
  (void)hipFuncSetAttribute(reinterpret_cast<const void*>(&attn_flash<32>),
                            hipFuncAttributeMaxDynamicSharedMemorySize, (int)lds32);
  {
    dim3 g(S / 32, H, Bb);
    attn_flash<64><<<g, 64, lds64, stream>>>(Qbf, Kbf, Vbf, impM, ctx_bf, S, H,
                                             0.125f);
  }
  {
    dim3 g(S / 32, THh, Bb);
    attn_flash<32><<<g, 64, lds32, stream>>>(Qt, Kt, Vt, impT, ctxT_bf, S, THh,
                                             0.17677669529f);
  }

  // 6) output projections: x1 = x + ctx@ow + ob ; tf = ctxT@ta_ow + ta_ob
  GEMM(0, true, true, true, true, ctx_bf, owT, D, sa_ob, x, x1, x1_bf, D, D);
  GEMM(0, true, false, true, true, ctxT_bf, towT, TD, ta_ob, nullptr, tf, tf_bf, TD, TD);

  // 7) tproj; gate = sigmoid([x1, tf] @ tg_w + tg_b) as two accumulated GEMMs
  GEMM(0, true, false, true, false, tf_bf, tpwT, TD, tp_b, nullptr, tproj, nullptr, D, TD);
  GEMM(0, true, false, true, false, x1_bf, tgwT, D + TD, tg_b, nullptr, gate, nullptr, D, D);
  GEMM(2, false, true, true, false, tf_bf, tgwT + D, D + TD, nullptr, gate, gate, nullptr, D, TD);

  // 8) x2 = LN(x1 + gate*tproj); xn3 = LN(x2) (reuse xn_bf)
  gated_add_ln<<<R, 256, 0, stream>>>(x1, gate, tproj, n2_g, n2_b, x2, D);
  layernorm_to_bf16<<<R, 256, 0, stream>>>(x2, n3_g, n3_b, xn_bf, D);

  // 9) FFN
  GEMM(1, true, false, false, true, xn_bf, ffw1T, D, ff_b1, nullptr, nullptr, ffh_bf, FF, D);
  GEMM(0, true, true, true, false, ffh_bf, ffw2T, FF, ff_b2, x2, (float*)d_out, nullptr, D, FF);

#undef GEMM
}
